// MultiHeadAttention_87522843558381
// MI455X (gfx1250) — compile-verified, matrix operand order checked by reading, not run on hardware
//
#include <hip/hip_runtime.h>
#include <hip/hip_bf16.h>

#define NH 8
#define DK 64
#define DM 512
#define BB 2
#define SS 4096
#define MT (BB * SS)  // 8192 rows total

typedef __attribute__((ext_vector_type(16))) _Float16 v16h;
typedef __attribute__((ext_vector_type(8)))  float    v8f;
typedef __attribute__((ext_vector_type(4)))  float    f32x4;
typedef __attribute__((ext_vector_type(4)))  _Float16 f16x4;
typedef __attribute__((ext_vector_type(4)))  unsigned int u32x4;

union FragU { v16h v; u32x4 q[2]; };
union PackU { f16x4 h[2]; u32x4 q; };

// Truncate a generic pointer to a __shared__ object to its LDS byte offset
// (ISA: LDS aperture address low 32 bits == LDS offset).
static __device__ inline unsigned lds_off(const void* p) {
  return (unsigned)(unsigned long long)p;
}

// CDNA5 LDS-DMA: copy 16 bytes per lane, global -> LDS, bypassing VGPRs.
// Tracked by ASYNCcnt; pair with s_wait_asynccnt + barrier before consuming.
static __device__ inline void async_ld_b128(unsigned lds_addr, const void* gaddr) {
  asm volatile("global_load_async_to_lds_b128 %0, %1, off"
               :: "v"(lds_addr), "v"((unsigned long long)gaddr) : "memory");
}

// Load a 16x32 f16 fragment (A or B operand) from a row-major tile with
// leading dimension `ld` (in halves, multiple of 8, base 16B aligned).
// Lane layout per CDNA5 ISA 16-bit A 16x32: lanes 0-15 row=lane, K chunks
// {g*8..g*8+7, 16+g*8..16+g*8+7} with g = lane>>4.
static __device__ inline v16h load_frag_f16(const _Float16* __restrict__ tile, int ld) {
  const int lane = threadIdx.x & 31;
  const int g = lane >> 4;
  const int r = lane & 15;
  const _Float16* p = tile + r * ld + g * 8;
  FragU f;
  f.q[0] = *reinterpret_cast<const u32x4*>(p);
  f.q[1] = *reinterpret_cast<const u32x4*>(p + 16);
  return f.v;
}

static __device__ inline v8f wmma_f16(v16h a, v16h b, v8f c) {
  return __builtin_amdgcn_wmma_f32_16x16x32_f16(false, a, false, b, (short)0, c, false, false);
}

// ---------------------------------------------------------------- converts
__global__ void cvt_f32_to_f16(const float* __restrict__ in, _Float16* __restrict__ out, int n4) {
  int i = blockIdx.x * blockDim.x + threadIdx.x;
  if (i < n4) {
    f32x4 x = reinterpret_cast<const f32x4*>(in)[i];
    reinterpret_cast<f16x4*>(out)[i] = __builtin_convertvector(x, f16x4);
  }
}

// ---------------------------------------------------------------- GEMM  y = X @ W^T + b
// X: [M=8192, K=512] f16 row-major.  W: [N=512, K=512] f16 row-major.
// mode 0: out f16 head-split [b][h][s][dk]
// mode 1: out f16 head-split transposed [b][h][dk][s]
// mode 2: out f32 flat [m][512]
// Double-buffered LDS staging via async LDS-DMA (3 x b128 per thread per tile).
__global__ __launch_bounds__(256) void gemm_xwT(const _Float16* __restrict__ X,
                                                const _Float16* __restrict__ W,
                                                const float* __restrict__ bias,
                                                void* __restrict__ outp, int mode) {
  __shared__ __align__(16) _Float16 As[2][128 * 32];
  __shared__ __align__(16) _Float16 Bs[2][64 * 32];
  const int m0 = blockIdx.x * 128;
  const int n0 = blockIdx.y * 64;
  const int t = threadIdx.x;
  const int wave = t >> 5;
  const int wm = (wave >> 1) * 32;   // wave covers 32 rows
  const int wn = (wave & 1) * 32;    // and 32 cols
  v8f acc[2][2] = {{{}, {}}, {{}, {}}};

  auto issue_tile = [&](int k0, int buf) {
    // A tile 128x32: 2 chunks of 8 halves per thread
    #pragma unroll
    for (int c = t * 2; c < t * 2 + 2; ++c) {
      int row = c >> 2, off = (c & 3) * 8;
      async_ld_b128(lds_off(&As[buf][row * 32 + off]),
                    &X[(size_t)(m0 + row) * DM + k0 + off]);
    }
    // B tile 64x32: 1 chunk per thread
    int row = t >> 2, off = (t & 3) * 8;
    async_ld_b128(lds_off(&Bs[buf][row * 32 + off]),
                  &W[(size_t)(n0 + row) * DM + k0 + off]);
  };

  issue_tile(0, 0);
  const int nk = DM / 32;  // 16
  for (int it = 0; it < nk; ++it) {
    if (it + 1 < nk) {
      issue_tile((it + 1) * 32, (it + 1) & 1);
      asm volatile("s_wait_asynccnt 0x3" ::: "memory");  // current tile landed
    } else {
      asm volatile("s_wait_asynccnt 0x0" ::: "memory");
    }
    __syncthreads();
    const _Float16* Ab = As[it & 1];
    const _Float16* Bb = Bs[it & 1];
    v16h a0 = load_frag_f16(Ab + (wm + 0) * 32, 32);
    v16h a1 = load_frag_f16(Ab + (wm + 16) * 32, 32);
    v16h b0 = load_frag_f16(Bb + (wn + 0) * 32, 32);
    v16h b1 = load_frag_f16(Bb + (wn + 16) * 32, 32);
    acc[0][0] = wmma_f16(a0, b0, acc[0][0]);
    acc[0][1] = wmma_f16(a0, b1, acc[0][1]);
    acc[1][0] = wmma_f16(a1, b0, acc[1][0]);
    acc[1][1] = wmma_f16(a1, b1, acc[1][1]);
    __syncthreads();  // all waves done with buf (it&1) before it is overwritten
  }

  const int lane = t & 31;
  const int g = lane >> 4, cn = lane & 15;
  #pragma unroll
  for (int i = 0; i < 2; ++i)
    #pragma unroll
    for (int j = 0; j < 2; ++j)
      #pragma unroll
      for (int e = 0; e < 8; ++e) {
        int m = m0 + wm + i * 16 + 8 * g + e;
        int n = n0 + wn + j * 16 + cn;
        float val = acc[i][j][e] + bias[n];
        if (mode == 2) {
          reinterpret_cast<float*>(outp)[(size_t)m * DM + n] = val;
        } else {
          int b = m >> 12, s = m & (SS - 1);
          int h = n >> 6, d = n & (DK - 1);
          size_t bh = (size_t)(b * NH + h);
          if (mode == 0)
            reinterpret_cast<_Float16*>(outp)[(bh * SS + s) * DK + d] = (_Float16)val;
          else
            reinterpret_cast<_Float16*>(outp)[(bh * DK + d) * SS + s] = (_Float16)val;
        }
      }
}

// ---------------------------------------------------------------- scores = q k^T / sqrt(dk), causal mask
__global__ __launch_bounds__(256) void scores_kernel(const _Float16* __restrict__ qh,
                                                     const _Float16* __restrict__ kh,
                                                     float* __restrict__ attn) {
  const int qt = blockIdx.x * 128;
  const int kt = blockIdx.y * 128;
  const int bh = blockIdx.z;
  float* __restrict__ outp = attn + (size_t)bh * SS * SS;

  if (blockIdx.y > blockIdx.x) {  // fully masked tile: no GEMM, vector fill
    const f32x4 neg = {-1e9f, -1e9f, -1e9f, -1e9f};
    for (int c = threadIdx.x; c < 128 * 128 / 4; c += 256) {
      int row = c >> 5, col = (c & 31) * 4;
      *reinterpret_cast<f32x4*>(&outp[(size_t)(qt + row) * SS + kt + col]) = neg;
    }
    return;
  }

  __shared__ __align__(16) _Float16 Qs[128 * 64];
  __shared__ __align__(16) _Float16 Ks[128 * 64];
  const _Float16* qb = qh + (size_t)bh * SS * DK;
  const _Float16* kb = kh + (size_t)bh * SS * DK;
  const int t = threadIdx.x;

  // stage 128x64 q and k tiles (full d_k) via async LDS-DMA: 8 chunks/thread
  #pragma unroll
  for (int c = t * 4; c < t * 4 + 4; ++c) {
    int row = c >> 3, off = (c & 7) * 8;
    async_ld_b128(lds_off(&Qs[row * 64 + off]), &qb[(size_t)(qt + row) * DK + off]);
    async_ld_b128(lds_off(&Ks[row * 64 + off]), &kb[(size_t)(kt + row) * DK + off]);
  }
  asm volatile("s_wait_asynccnt 0x0" ::: "memory");
  __syncthreads();

  const int wave = t >> 5;
  const int wqm = (wave >> 1) * 32;  // 32 q rows per wave
  const int wkn = (wave & 1) * 64;   // 64 k cols per wave
  v8f acc[2][4] = {{{}, {}, {}, {}}, {{}, {}, {}, {}}};

  #pragma unroll
  for (int d0 = 0; d0 < DK; d0 += 32) {
    v16h a0 = load_frag_f16(Qs + (wqm + 0) * 64 + d0, 64);
    v16h a1 = load_frag_f16(Qs + (wqm + 16) * 64 + d0, 64);
    #pragma unroll
    for (int j = 0; j < 4; ++j) {
      v16h b = load_frag_f16(Ks + (wkn + j * 16) * 64 + d0, 64);
      acc[0][j] = wmma_f16(a0, b, acc[0][j]);
      acc[1][j] = wmma_f16(a1, b, acc[1][j]);
    }
  }

  const int lane = t & 31;
  const int g = lane >> 4, cn = lane & 15;
  #pragma unroll
  for (int i = 0; i < 2; ++i)
    #pragma unroll
    for (int j = 0; j < 4; ++j)
      #pragma unroll
      for (int e = 0; e < 8; ++e) {
        int q = qt + wqm + i * 16 + 8 * g + e;
        int k = kt + wkn + j * 16 + cn;
        float val = acc[i][j][e] * 0.125f;  // 1/sqrt(64)
        val = (k > q) ? -1e9f : val;
        outp[(size_t)q * SS + k] = val;
      }
}

// ---------------------------------------------------------------- row softmax, one wave32 per row
__global__ __launch_bounds__(256) void softmax_kernel(float* __restrict__ attn) {
  const int wave = threadIdx.x >> 5;
  const int lane = threadIdx.x & 31;
  const size_t row = (size_t)blockIdx.x * 8 + wave;
  float* __restrict__ p = attn + row * SS;

  float m = -3.4e38f, s = 0.0f;
  for (int i = lane; i < SS; i += 32) {
    float x = p[i];
    float nm = fmaxf(m, x);
    s = s * __expf(m - nm) + __expf(x - nm);
    m = nm;
  }
  #pragma unroll
  for (int off = 16; off > 0; off >>= 1) {
    float om = __shfl_xor(m, off, 32);
    float os = __shfl_xor(s, off, 32);
    float nm = fmaxf(m, om);
    s = s * __expf(m - nm) + os * __expf(om - nm);
    m = nm;
  }
  float inv = 1.0f / s;
  for (int i = lane; i < SS; i += 32) p[i] = __expf(p[i] - m) * inv;
}

// ---------------------------------------------------------------- ctx = attn @ v (v pre-transposed)
// Double-buffered: vT tile via async LDS-DMA, attn tile via vectorized f32->f16 convert.
__global__ __launch_bounds__(256) void av_kernel(const float* __restrict__ attn,
                                                 const _Float16* __restrict__ vT,
                                                 _Float16* __restrict__ ctx16) {
  __shared__ __align__(16) _Float16 As[2][128 * 32];
  __shared__ __align__(16) _Float16 Bs[2][64 * 32];
  const int qt = blockIdx.x * 128;
  const int bh = blockIdx.y;
  const int b = bh / NH, h = bh % NH;
  const float* __restrict__ A = attn + (size_t)bh * SS * SS;
  const _Float16* __restrict__ Vt = vT + (size_t)bh * DK * SS;
  const int t = threadIdx.x;
  const int wave = t >> 5;
  const int wm = (wave >> 1) * 32;
  const int wn = (wave & 1) * 32;
  v8f acc[2][2] = {{{}, {}}, {{}, {}}};

  auto issue_b = [&](int k0, int buf) {  // vT tile 64x32, 1 b128 per thread
    int row = t >> 2, off = (t & 3) * 8;
    async_ld_b128(lds_off(&Bs[buf][row * 32 + off]),
                  &Vt[(size_t)row * SS + k0 + off]);
  };
  auto load_a = [&](int k0, int buf) {  // attn tile 128x32, f32 -> f16
    int row = t >> 1, c0 = (t & 1) * 16;
    const f32x4* src = reinterpret_cast<const f32x4*>(&A[(size_t)(qt + row) * SS + k0 + c0]);
    PackU p0, p1;
    p0.h[0] = __builtin_convertvector(src[0], f16x4);
    p0.h[1] = __builtin_convertvector(src[1], f16x4);
    p1.h[0] = __builtin_convertvector(src[2], f16x4);
    p1.h[1] = __builtin_convertvector(src[3], f16x4);
    *reinterpret_cast<u32x4*>(&As[buf][row * 32 + c0]) = p0.q;
    *reinterpret_cast<u32x4*>(&As[buf][row * 32 + c0 + 8]) = p1.q;
  };

  const int nk = (qt + 128) >> 5;  // causal: columns beyond the q tile never contribute
  issue_b(0, 0);
  load_a(0, 0);
  for (int it = 0; it < nk; ++it) {
    if (it + 1 < nk) {
      issue_b((it + 1) * 32, (it + 1) & 1);
      asm volatile("s_wait_asynccnt 0x1" ::: "memory");  // current vT tile landed
      load_a((it + 1) * 32, (it + 1) & 1);
    } else {
      asm volatile("s_wait_asynccnt 0x0" ::: "memory");
    }
    __syncthreads();
    const _Float16* Ab = As[it & 1];
    const _Float16* Bb = Bs[it & 1];
    v16h a0 = load_frag_f16(Ab + (wm + 0) * 32, 32);
    v16h a1 = load_frag_f16(Ab + (wm + 16) * 32, 32);
    v16h b0 = load_frag_f16(Bb + (wn + 0) * 32, 32);
    v16h b1 = load_frag_f16(Bb + (wn + 16) * 32, 32);
    acc[0][0] = wmma_f16(a0, b0, acc[0][0]);
    acc[0][1] = wmma_f16(a0, b1, acc[0][1]);
    acc[1][0] = wmma_f16(a1, b0, acc[1][0]);
    acc[1][1] = wmma_f16(a1, b1, acc[1][1]);
    __syncthreads();
  }

  const int lane = t & 31;
  const int g = lane >> 4, cn = lane & 15;
  #pragma unroll
  for (int i = 0; i < 2; ++i)
    #pragma unroll
    for (int j = 0; j < 2; ++j)
      #pragma unroll
      for (int e = 0; e < 8; ++e) {
        int q = qt + wm + i * 16 + 8 * g + e;
        int d = wn + j * 16 + cn;
        ctx16[((size_t)b * SS + q) * DM + h * DK + d] = (_Float16)acc[i][j][e];
      }
}

// ---------------------------------------------------------------- host launcher
extern "C" void kernel_launch(void* const* d_in, const int* in_sizes, int n_in,
                              void* d_out, int out_size, void* d_ws, size_t ws_size,
                              hipStream_t stream) {
  (void)in_sizes; (void)n_in; (void)out_size; (void)ws_size;
  const float* Q  = (const float*)d_in[0];
  const float* K  = (const float*)d_in[1];
  const float* V  = (const float*)d_in[2];
  const float* Wq = (const float*)d_in[3];
  const float* bq = (const float*)d_in[4];
  const float* Wk = (const float*)d_in[5];
  const float* bk = (const float*)d_in[6];
  const float* Wv = (const float*)d_in[7];
  const float* bv = (const float*)d_in[8];
  const float* Wo = (const float*)d_in[9];
  const float* bo = (const float*)d_in[10];

  const size_t NX = (size_t)MT * DM;   // 4,194,304
  const size_t NW = (size_t)DM * DM;   //   262,144

  _Float16* Xq16  = (_Float16*)d_ws;
  _Float16* Xk16  = Xq16 + NX;
  _Float16* Xv16  = Xk16 + NX;
  _Float16* Wq16  = Xv16 + NX;
  _Float16* Wk16  = Wq16 + NW;
  _Float16* Wv16  = Wk16 + NW;
  _Float16* Wo16  = Wv16 + NW;
  _Float16* qh    = Wo16 + NW;                 // [b][h][s][dk]
  _Float16* kh    = qh + NX;                   // [b][h][s][dk]
  _Float16* vT    = kh + NX;                   // [b][h][dk][s]
  _Float16* ctx16 = vT + NX;                   // [b][s][dm]

  float* outF = (float*)d_out;                 // [B,S,DM]
  float* attn = outF + NX;                     // [B,H,S,S]

  // fp32 -> f16 converts
  cvt_f32_to_f16<<<(int)(NX / 4 + 255) / 256, 256, 0, stream>>>(Q, Xq16, (int)(NX / 4));
  cvt_f32_to_f16<<<(int)(NX / 4 + 255) / 256, 256, 0, stream>>>(K, Xk16, (int)(NX / 4));
  cvt_f32_to_f16<<<(int)(NX / 4 + 255) / 256, 256, 0, stream>>>(V, Xv16, (int)(NX / 4));
  cvt_f32_to_f16<<<(int)(NW / 4 + 255) / 256, 256, 0, stream>>>(Wq, Wq16, (int)(NW / 4));
  cvt_f32_to_f16<<<(int)(NW / 4 + 255) / 256, 256, 0, stream>>>(Wk, Wk16, (int)(NW / 4));
  cvt_f32_to_f16<<<(int)(NW / 4 + 255) / 256, 256, 0, stream>>>(Wv, Wv16, (int)(NW / 4));
  cvt_f32_to_f16<<<(int)(NW / 4 + 255) / 256, 256, 0, stream>>>(Wo, Wo16, (int)(NW / 4));

  // projections
  dim3 gGemm(MT / 128, DM / 64);
  gemm_xwT<<<gGemm, 256, 0, stream>>>(Xq16, Wq16, bq, qh, 0);
  gemm_xwT<<<gGemm, 256, 0, stream>>>(Xk16, Wk16, bk, kh, 0);
  gemm_xwT<<<gGemm, 256, 0, stream>>>(Xv16, Wv16, bv, vT, 1);

  // scores + softmax + attn@v
  dim3 gScores(SS / 128, SS / 128, BB * NH);
  scores_kernel<<<gScores, 256, 0, stream>>>(qh, kh, attn);
  softmax_kernel<<<(BB * NH * SS) / 8, 256, 0, stream>>>(attn);
  dim3 gAv(SS / 128, BB * NH);
  av_kernel<<<gAv, 256, 0, stream>>>(attn, vT, ctx16);

  // output projection (fp32 out)
  gemm_xwT<<<gGemm, 256, 0, stream>>>(ctx16, Wo16, bo, (void*)outF, 2);
}
